// FLAMEGeneratorV9_91293824844456
// MI455X (gfx1250) — matrix operand order repaired and checked by reading, not compiled, not run on hardware
//
#include <hip/hip_runtime.h>
#include <hip/hip_bf16.h>
#include <math.h>

typedef __bf16 bf16;
typedef bf16  v16bf __attribute__((ext_vector_type(16)));
typedef bf16  v8bf  __attribute__((ext_vector_type(8)));
typedef float v8f   __attribute__((ext_vector_type(8)));
typedef unsigned int u32x4 __attribute__((ext_vector_type(4)));
typedef int i32x4 __attribute__((ext_vector_type(4)));
typedef int i32x8 __attribute__((ext_vector_type(8)));

#define DEVINL __device__ __forceinline__

#if defined(__has_builtin)
#if __has_builtin(__builtin_amdgcn_tensor_load_to_lds)
#define HAVE_TDM 1
#endif
#endif

constexpr int B_ = 16, T_ = 4096;

DEVINL float geluf(float x) { return 0.5f * x * (1.0f + erff(x * 0.70710678118654752f)); }
DEVINL float sigm(float x)  { return 1.0f / (1.0f + expf(-x)); }

// Async copy of 16B global -> LDS (ASYNCcnt-tracked, per-lane).
DEVINL void async_b128(uint32_t lds_off, const void* gaddr) {
  asm volatile("global_load_async_to_lds_b128 %0, %1, off"
               :: "v"(lds_off), "v"((uint64_t)gaddr) : "memory");
}
DEVINL uint32_t lds_off_of(const void* p) { return (uint32_t)(size_t)p; }

#ifdef HAVE_TDM
// TDM: DMA a contiguous run of nelem fp32 from global to LDS (TENSORcnt).
// D# group0: count=1 | lds_addr | global_addr(57b) | type=2.
// D# group1: data_size=4B; tensor_dim0 = tile_dim0 = nelem; tensor_dim1 = 1.
// Remaining groups zero (1D tile).  amdgpu-toolchain 6-arg builtin form.
DEVINL void tdm_load_1d(uint32_t lds_addr, const void* gaddr, int nelem) {
  uint64_t ga = (uint64_t)gaddr;
  u32x4 g0;
  g0[0] = 1u;                                                  // count=1
  g0[1] = lds_addr;
  g0[2] = (uint32_t)ga;
  g0[3] = (uint32_t)((ga >> 32) & 0x1FFFFFFu) | (2u << 30);    // addr hi | type=2
  i32x8 g1;
  g1[0] = (2 << 16);                                           // data_size = 4B
  g1[1] = (int)((nelem & 0xFFFF) << 16);                       // tensor_dim0 lo
  g1[2] = (int)((1 << 16) | ((nelem >> 16) & 0xFFFF));         // dim1=1 | dim0 hi
  g1[3] = (int)((nelem & 0xFFFF) << 16);                       // tile_dim0 | dim1 hi
  g1[4] = 0;
  g1[5] = nelem;                                               // dim0 stride
  g1[6] = 0;
  g1[7] = 0;
  i32x4 z4 = {0, 0, 0, 0};
  i32x8 z8 = {0, 0, 0, 0, 0, 0, 0, 0};
  __builtin_amdgcn_tensor_load_to_lds(g0, g1, z4, z4, z8, 0);
}
#endif

// Block-wide sum over 192 threads (inactive threads must pass 0).
DEVINL float blk_sum_192(float v, float* s_red, int tid) {
  s_red[tid] = v; __syncthreads();
  if (tid < 64) s_red[tid] += s_red[tid + 128]; __syncthreads();
  if (tid < 64) s_red[tid] += s_red[tid + 64];  __syncthreads();
  if (tid < 32) s_red[tid] += s_red[tid + 32];  __syncthreads();
  if (tid < 16) s_red[tid] += s_red[tid + 16];  __syncthreads();
  if (tid < 8)  s_red[tid] += s_red[tid + 8];   __syncthreads();
  if (tid < 4)  s_red[tid] += s_red[tid + 4];   __syncthreads();
  if (tid < 2)  s_red[tid] += s_red[tid + 2];   __syncthreads();
  if (tid == 0) s_red[0]   += s_red[1];         __syncthreads();
  float r = s_red[0]; __syncthreads();
  return r;
}

// ---------------------------------------------------------------------------
// Pack a row-major fp32 weight [K,N] into bf16 WMMA B-matrix tiles.
// ---------------------------------------------------------------------------
__global__ void pack_b(const float* __restrict__ W, bf16* __restrict__ dst, int K, int N) {
  const int KT = K >> 5;
  const size_t total = (size_t)K * N;
  for (size_t idx = (size_t)blockIdx.x * 256 + threadIdx.x; idx < total;
       idx += (size_t)gridDim.x * 256) {
    int tile = (int)(idx >> 9);
    int w    = (int)(idx & 511);
    int lane = w >> 4, e = w & 15;
    int nt = tile / KT, kt = tile - nt * KT;
    int n = nt * 16 + (lane & 15);
    int k = kt * 32 + (e >> 3) * 16 + (lane >> 4) * 8 + (e & 7);
    dst[idx] = (bf16)W[(size_t)k * N + n];
  }
}

__global__ void pack_hbias(const float* e, const float* j, const float* y,
                           const float* o, float* dst) {
  int t = threadIdx.x;
  if (t < 192)      dst[t] = e[t];
  else if (t < 256) dst[t] = j[t - 192];
  else if (t < 320) dst[t] = y[t - 256];
  else              dst[t] = o[t - 320];
}

// ---------------------------------------------------------------------------
// WMMA GEMM: C[M,N] = epilogue(A[M,K] @ Bpacked + bias).
// 4 waves/block -> 64x64 C tile.  B k-panel (4KB) async-staged to LDS with
// double buffering; A fragments software-pipelined one k-step ahead.
// grid = (M/64, N/64).
// ---------------------------------------------------------------------------
__global__ __launch_bounds__(128)
void wmma_gemm(const bf16* __restrict__ A, int lda,
               const bf16* __restrict__ Bp,
               const float* __restrict__ bias,
               float* __restrict__ C, int ldc,
               int K, int gelu_ep) {
  __shared__ __align__(16) bf16 sB[2][2048];   // 2 x 4KB k-panels
  const int tid  = threadIdx.x;
  const int wave = tid >> 5, lane = tid & 31;
  const int half = lane >> 4, nlo = lane & 15;
  const int KT = K >> 5;
  const int nq = blockIdx.y;
  const bf16* aptr = A + ((size_t)blockIdx.x * 64 + wave * 16 + nlo) * (size_t)lda;

  // thread tid stages 32B of the 4KB panel: halfs [tid*16, tid*16+16)
  const int p  = tid * 16;
  const int pq = p >> 9, poff = p & 511;
  const bf16* psrc0 = Bp + ((size_t)(nq * 4 + pq) * KT) * 512 + poff;

  auto stage = [&](int buf, int kt) {
    const bf16* src = psrc0 + (size_t)kt * 512;
    uint32_t dst = lds_off_of(&sB[buf][p]);
    async_b128(dst,      src);
    async_b128(dst + 16, src + 8);
  };

  v8f acc[4];
  acc[0] = {}; acc[1] = {}; acc[2] = {}; acc[3] = {};

  stage(0, 0);
  v8bf a0 = *(const v8bf*)(aptr + half * 8);
  v8bf a1 = *(const v8bf*)(aptr + 16 + half * 8);

  for (int kt = 0; kt < KT; ++kt) {
    const int buf = kt & 1;
    if (kt + 1 < KT) {
      stage(buf ^ 1, kt + 1);
      asm volatile("s_wait_asynccnt 0x2" ::: "memory");
    } else {
      asm volatile("s_wait_asynccnt 0x0" ::: "memory");
    }
    __syncthreads();   // panel kt visible to all waves

    v16bf a = __builtin_shufflevector(a0, a1, 0,1,2,3,4,5,6,7,8,9,10,11,12,13,14,15);
    v8bf n0 = {}, n1 = {};
    if (kt + 1 < KT) {  // issue next A loads before this step's WMMAs
      const int k1 = (kt + 1) * 32 + half * 8;
      n0 = *(const v8bf*)(aptr + k1);
      n1 = *(const v8bf*)(aptr + k1 + 16);
    }
#pragma unroll
    for (int q = 0; q < 4; ++q) {
      v16bf b = *(const v16bf*)(sB[buf] + q * 512 + lane * 16);
      acc[q] = __builtin_amdgcn_wmma_f32_16x16x32_bf16(
          false, a, false, b, (short)0, acc[q], false, false);
    }
    __syncthreads();   // all waves done with panel before it is re-staged
    a0 = n0; a1 = n1;
  }

  const size_t r0 = (size_t)blockIdx.x * 64 + wave * 16;
#pragma unroll
  for (int q = 0; q < 4; ++q) {
    int n = (nq * 4 + q) * 16 + nlo;
    float bv = bias ? bias[n] : 0.0f;
#pragma unroll
    for (int v = 0; v < 8; ++v) {
      float r = acc[q][v] + bv;
      if (gelu_ep) r = geluf(r);
      C[(r0 + v + half * 8) * (size_t)ldc + n] = r;
    }
  }
}

// ---------------------------------------------------------------------------
// Bidirectional GRU scan, one workgroup per direction (blockIdx.x = dir).
// LDS: packed Whh (221184B) + h fp32 (12288B) + h A-fragments (6144B) +
// double-buffered gi step blocks (2 x 36864B) = 313344B  (<320KB/WGP).
// gi[t+1] is DMA'd by the Tensor Data Mover (one descriptor per step, issued
// by wave 0, TENSORcnt-tracked) while step t computes its 36 WMMAs/wave.
// ---------------------------------------------------------------------------
__global__ __launch_bounds__(192)
void gru_scan(const bf16* __restrict__ WhhF, const bf16* __restrict__ WhhB,
              const float* __restrict__ bhhF, const float* __restrict__ bhhB,
              const float* __restrict__ giF,  const float* __restrict__ giB,
              bf16* __restrict__ ybf, float* __restrict__ yf, int T) {
  extern __shared__ char smem[];
  bf16*  sWhh = (bf16*)smem;                          // 110592 halfs
  float* sH   = (float*)(smem + 221184);              // 16*192 fp32
  bf16*  sHA  = (bf16*)(smem + 221184 + 12288);       // 6 A-tiles * 512 halfs
  float* sGI0 = (float*)(smem + 239616);              // 9216 fp32
  float* sGI1 = (float*)(smem + 239616 + 36864);      // 9216 fp32

  const int dir = blockIdx.x;
  const bf16*  Whh = dir ? WhhB : WhhF;
  const float* bhh = dir ? bhhB : bhhF;
  const float* gi  = dir ? giB  : giF;
  const int colOff = dir ? 192 : 0;

  const int tid = threadIdx.x;
  const int wave = tid >> 5, lane = tid & 31;
  const int half = lane >> 4, nlo = lane & 15;
  const int j0 = wave * 32;

  { // stage packed Whh into LDS
    const uint4* s = (const uint4*)Whh;
    uint4* d = (uint4*)sWhh;
    for (int i = tid; i < 13824; i += 192) d[i] = s[i];
  }
  for (int i = tid; i < 3072; i += 192) { sH[i] = 0.0f; sHA[i] = (bf16)0.0f; }

  const uint32_t gdst0 = lds_off_of(sGI0);
  const uint32_t gdst1 = lds_off_of(sGI1);
#ifdef HAVE_TDM
  auto stageGI = [&](int buf, int tstep) {
    if (wave == 0 && lane == 0)
      tdm_load_1d(buf ? gdst1 : gdst0, gi + (size_t)tstep * 9216, 9216);
  };
#else
  auto stageGI = [&](int buf, int tstep) {  // fallback: per-lane async copies
    const char* src = (const char*)(gi + (size_t)tstep * 9216) + tid * 192;
    uint32_t dst = (buf ? gdst1 : gdst0) + (uint32_t)tid * 192;
#pragma unroll
    for (int i = 0; i < 12; ++i)
      async_b128(dst + i * 16, src + i * 16);
  };
#endif

  stageGI(0, dir ? T - 1 : 0);

  float bh[2][3];
#pragma unroll
  for (int s = 0; s < 2; ++s) {
    int col = j0 + s * 16 + nlo;
    bh[s][0] = bhh[col]; bh[s][1] = bhh[192 + col]; bh[s][2] = bhh[384 + col];
  }

  for (int t = 0; t < T; ++t) {
    const int tt = dir ? (T - 1 - t) : t;
    const float* sGI = (t & 1) ? sGI1 : sGI0;
    if (t + 1 < T) {
      stageGI((t + 1) & 1, dir ? (T - 2 - t) : (t + 1));
#ifdef HAVE_TDM
      if (wave == 0) __builtin_amdgcn_s_wait_tensorcnt(1);
#else
      asm volatile("s_wait_asynccnt 0xc" ::: "memory");
#endif
    } else {
#ifdef HAVE_TDM
      if (wave == 0) __builtin_amdgcn_s_wait_tensorcnt(0);
#else
      asm volatile("s_wait_asynccnt 0x0" ::: "memory");
#endif
    }
    if (t + 2 < T && tid < 144) { // warm L2 for step t+2
      const int t3 = dir ? (T - 3 - t) : (t + 2);
      __builtin_prefetch((const char*)(gi + (size_t)t3 * 9216) + tid * 256, 0, 1);
    }
    __syncthreads();  // gi[t] staged; h updates from t-1 visible

    v16bf aF[6];
#pragma unroll
    for (int kt = 0; kt < 6; ++kt)
      aF[kt] = *(const v16bf*)(sHA + kt * 512 + lane * 16);

    v8f acc[3][2];
#pragma unroll
    for (int g = 0; g < 3; ++g)
#pragma unroll
      for (int s = 0; s < 2; ++s) {
        v8f c = {};
        const int nt = g * 12 + wave * 2 + s;
#pragma unroll
        for (int kt = 0; kt < 6; ++kt) {
          v16bf b = *(const v16bf*)(sWhh + (size_t)(nt * 6 + kt) * 512 + lane * 16);
          c = __builtin_amdgcn_wmma_f32_16x16x32_bf16(
              false, aF[kt], false, b, (short)0, c, false, false);
        }
        acc[g][s] = c;
      }

    float hnew[2][8];
#pragma unroll
    for (int s = 0; s < 2; ++s) {
      const int col = j0 + s * 16 + nlo;
#pragma unroll
      for (int v = 0; v < 8; ++v) {
        const int m = v + half * 8;
        const float* gr = sGI + m * 576;
        float r  = sigm(gr[col]       + acc[0][s][v] + bh[s][0]);
        float z  = sigm(gr[192 + col] + acc[1][s][v] + bh[s][1]);
        float hn = acc[2][s][v] + bh[s][2];
        float n  = tanhf(gr[384 + col] + r * hn);
        float hp = sH[m * 192 + col];
        hnew[s][v] = (1.0f - z) * n + z * hp;
      }
    }
    __syncthreads();  // all reads of sH/sGI done before overwrite
#pragma unroll
    for (int s = 0; s < 2; ++s) {
      const int col = j0 + s * 16 + nlo;
      const int kt = col >> 5, kk = col & 31;
      const int laHi = (kk >> 3) & 1;
      const int e = (kk & 7) + ((kk & 16) ? 8 : 0);
#pragma unroll
      for (int v = 0; v < 8; ++v) {
        const int m = v + half * 8;
        float hv = hnew[s][v];
        sH[m * 192 + col] = hv;
        sHA[kt * 512 + (m | (laHi << 4)) * 16 + e] = (bf16)hv;
        const size_t oy = ((size_t)tt * 16 + m) * 384 + colOff + col;
        if (ybf) ybf[oy] = (bf16)hv; else yf[oy] = hv;
      }
    }
  }
}

// ---------------------------------------------------------------------------
// Per-batch conditioning: emo MLP, text pooling/MLP, amp MLP, FiLM vectors.
// ---------------------------------------------------------------------------
__global__ __launch_bounds__(192)
void prep_small(const float* emotion, const int* token_ids, const float* token_mask,
                const float* phonf,
                const float* char_embed, const float* phon_W, const float* phon_b,
                const float* txt_W, const float* txt_b, const float* txt_g, const float* txt_bb,
                const float* emo_W1, const float* emo_b1, const float* emo_W2, const float* emo_b2,
                const float* amp_W1, const float* amp_b1, const float* amp_g, const float* amp_bb,
                const float* amp_W2, const float* amp_b2,
                const float* mW, const float* mb, const float* cW, const float* cb,
                const float* nW, const float* nb, const float* bW, const float* bb2,
                const float* amp_scale,
                const float* fes_W, const float* fes_b, const float* fet_W, const float* fet_b,
                const float* fts_W, const float* fts_b, const float* ftt_W, const float* ftt_b,
                float* femS, float* femT, float* ftxS, float* ftxT, float* amps) {
  __shared__ float s_in[10], s_ph[10], s_e1[64], s_tin[128], s_h[128];
  __shared__ float s_emo[192], s_txt[192], s_red[192];
  const int b = blockIdx.x, tid = threadIdx.x;

  if (tid < 10) { s_in[tid] = emotion[b * 10 + tid]; s_ph[tid] = phonf[b * 10 + tid]; }
  __syncthreads();

  if (tid < 64) {
    float a = emo_b1[tid];
    for (int i = 0; i < 10; ++i) a += s_in[i] * emo_W1[i * 64 + tid];
    s_e1[tid] = geluf(a);
  }
  __syncthreads();
  {
    float a = emo_b2[tid];
    for (int k = 0; k < 64; ++k) a += s_e1[k] * emo_W2[k * 192 + tid];
    s_emo[tid] = a;
  }

  if (tid < 64) {
    float num = 0.0f, den = 0.0f;
    for (int l = 0; l < 64; ++l) {
      float w = token_mask[b * 64 + l];
      den += w;
      num += w * char_embed[token_ids[b * 64 + l] * 64 + tid];
    }
    s_tin[tid] = num / fmaxf(den, 1.0f);
  } else if (tid < 128) {
    int j = tid - 64;
    float a = phon_b[j];
    for (int i = 0; i < 10; ++i) a += s_ph[i] * phon_W[i * 64 + j];
    s_tin[tid] = a;
  }
  __syncthreads();
  {
    float tv = txt_b[tid];
    for (int k = 0; k < 128; ++k) tv += s_tin[k] * txt_W[k * 192 + tid];
    float m = blk_sum_192(tv, s_red, tid) * (1.0f / 192.0f);
    float d = tv - m;
    float var = blk_sum_192(d * d, s_red, tid) * (1.0f / 192.0f);
    s_txt[tid] = geluf(d * rsqrtf(var + 1e-5f) * txt_g[tid] + txt_bb[tid]);
  }

  float ah = 0.0f;
  if (tid < 128) {
    ah = amp_b1[tid];
    for (int i = 0; i < 10; ++i) ah += s_in[i] * amp_W1[i * 128 + tid];
  }
  float am = blk_sum_192(tid < 128 ? ah : 0.0f, s_red, tid) * (1.0f / 128.0f);
  float ad = (tid < 128) ? (ah - am) : 0.0f;
  float av = blk_sum_192(ad * ad, s_red, tid) * (1.0f / 128.0f);
  if (tid < 128) s_h[tid] = geluf(ad * rsqrtf(av + 1e-5f) * amp_g[tid] + amp_bb[tid]);
  __syncthreads();
  float a2 = 0.0f;
  if (tid < 128) {
    a2 = amp_b2[tid];
    for (int k = 0; k < 128; ++k) a2 += s_h[k] * amp_W2[k * 128 + tid];
  }
  __syncthreads();
  if (tid < 128) s_h[tid] = geluf(a2);
  __syncthreads();

  float s = amp_scale[0];
  if (tid < 8) {
    float a = mb[tid];
    for (int k = 0; k < 128; ++k) a += s_h[k] * mW[k * 8 + tid];
    amps[b * 22 + tid] = tanhf(a) * s;
  } else if (tid < 12) {
    int j = tid - 8; float a = cb[j];
    for (int k = 0; k < 128; ++k) a += s_h[k] * cW[k * 4 + j];
    amps[b * 22 + tid] = tanhf(a) * s * 0.6f;
  } else if (tid < 16) {
    int j = tid - 12; float a = nb[j];
    for (int k = 0; k < 128; ++k) a += s_h[k] * nW[k * 4 + j];
    amps[b * 22 + tid] = tanhf(a) * s * 0.4f;
  } else if (tid < 22) {
    int j = tid - 16; float a = bb2[j];
    for (int k = 0; k < 128; ++k) a += s_h[k] * bW[k * 6 + j];
    amps[b * 22 + tid] = tanhf(a) * s * 0.75f;
  }

  { float a = fes_b[tid]; for (int k = 0; k < 192; ++k) a += s_emo[k] * fes_W[k * 192 + tid]; femS[b * 192 + tid] = a; }
  { float a = fet_b[tid]; for (int k = 0; k < 192; ++k) a += s_emo[k] * fet_W[k * 192 + tid]; femT[b * 192 + tid] = a; }
  { float a = fts_b[tid]; for (int k = 0; k < 192; ++k) a += s_txt[k] * fts_W[k * 192 + tid]; ftxS[b * 192 + tid] = a; }
  { float a = ftt_b[tid]; for (int k = 0; k < 192; ++k) a += s_txt[k] * ftt_W[k * 192 + tid]; ftxT[b * 192 + tid] = a; }
}

// ---------------------------------------------------------------------------
// nmf_feat = gelu(LN(concat(nmf,time)@W+b)); both FiLMs; bf16 x0 row r=t*16+b.
// ---------------------------------------------------------------------------
__global__ __launch_bounds__(192)
void nmf_film(const float* __restrict__ nmf, const float* __restrict__ tf,
              const float* __restrict__ W, const float* __restrict__ bias,
              const float* __restrict__ g, const float* __restrict__ bb,
              const float* __restrict__ femS, const float* __restrict__ femT,
              const float* __restrict__ ftxS, const float* __restrict__ ftxT,
              bf16* __restrict__ x0) {
  __shared__ float s_in[21], s_red[192];
  const size_t r = blockIdx.x;
  const int b = (int)(r & 15), t = (int)(r >> 4);
  const int tid = threadIdx.x;
  if (tid < 17) s_in[tid] = nmf[((size_t)b * T_ + t) * 17 + tid];
  else if (tid < 21) s_in[tid] = tf[((size_t)b * T_ + t) * 4 + (tid - 17)];
  __syncthreads();
  float a = bias[tid];
  for (int i = 0; i < 21; ++i) a += s_in[i] * W[i * 192 + tid];
  float m = blk_sum_192(a, s_red, tid) * (1.0f / 192.0f);
  float d = a - m;
  float v = blk_sum_192(d * d, s_red, tid) * (1.0f / 192.0f);
  float x = geluf(d * rsqrtf(v + 1e-5f) * g[tid] + bb[tid]);
  x = femS[b * 192 + tid] * x + femT[b * 192 + tid];
  x = ftxS[b * 192 + tid] * x + ftxT[b * 192 + tid];
  x0[r * 192 + tid] = (bf16)x;
}

__global__ __launch_bounds__(192)
void post_ln(const float* __restrict__ x2, const float* __restrict__ g,
             const float* __restrict__ b, bf16* __restrict__ xln) {
  __shared__ float s_red[192];
  const size_t r = blockIdx.x;
  const int tid = threadIdx.x;
  float v0 = x2[r * 384 + tid], v1 = x2[r * 384 + 192 + tid];
  float m = blk_sum_192(v0 + v1, s_red, tid) * (1.0f / 384.0f);
  float d0 = v0 - m, d1 = v1 - m;
  float var = blk_sum_192(d0 * d0 + d1 * d1, s_red, tid) * (1.0f / 384.0f);
  float is = rsqrtf(var + 1e-5f);
  xln[r * 384 + tid]       = (bf16)(d0 * is * g[tid] + b[tid]);
  xln[r * 384 + 192 + tid] = (bf16)(d1 * is * g[192 + tid] + b[192 + tid]);
}

// ---------------------------------------------------------------------------
// Output heads from hh = gelu(LN(x)@W1pack + b1pack)  ([expr192|jaw64|eye64|ori64]).
// ---------------------------------------------------------------------------
__global__ __launch_bounds__(128)
void final_heads(const float* __restrict__ hh, const float* __restrict__ nmf,
                 const float* __restrict__ amps,
                 const float* eW2, const float* eb2,
                 const float* jW2, const float* jb2, const float* jsc,
                 const float* yW2, const float* yb2, const float* ysc,
                 const float* oW2, const float* ob2, const float* osc,
                 const float* n2mW, const float* n2mb,
                 const float* n2jW, const float* n2jb,
                 float* __restrict__ out) {
  __shared__ float s_h[384], s_nm[17];
  const size_t r = blockIdx.x;
  const int b = (int)(r & 15), t = (int)(r >> 4);
  const int tid = threadIdx.x;
  for (int i = tid; i < 384; i += 128) s_h[i] = hh[r * 384 + i];
  if (tid < 17) s_nm[tid] = nmf[((size_t)b * T_ + t) * 17 + tid];
  __syncthreads();
  float res = 0.0f; int c = -1;
  if (tid < 100) {
    float a = eb2[tid];
    for (int k = 0; k < 192; ++k) a += s_h[k] * eW2[k * 100 + tid];
    if (tid < 16) {
      float nm = n2mb[tid];
      for (int i = 0; i < 17; ++i) nm += s_nm[i] * n2mW[i * 16 + tid];
      a += 0.35f * nm;
    }
    const int AI[22] = {10,11,12,13,14,15,5,3,16,17,20,21,18,19,22,23,30,31,32,33,34,35};
#pragma unroll
    for (int q = 0; q < 22; ++q)
      if (AI[q] == tid) a += amps[b * 22 + q];
    res = a; c = tid;
  } else if (tid < 103) {
    int j = tid - 100;
    float a = jb2[j];
    for (int k = 0; k < 64; ++k) a += s_h[192 + k] * jW2[k * 3 + j];
    float v = tanhf(a) * jsc[j];
    if (j == 0) {
      float nj = n2jb[0];
      for (int i = 0; i < 17; ++i) nj += s_nm[i] * n2jW[i];
      v = fminf(fmaxf(v + 0.25f * tanhf(nj), -0.45f), 0.45f);
    }
    res = v; c = 100 + j;
  } else if (tid < 109) {
    int j = tid - 103;
    float a = yb2[j];
    for (int k = 0; k < 64; ++k) a += s_h[256 + k] * yW2[k * 6 + j];
    res = tanhf(a) * ysc[j]; c = 103 + j;
  } else if (tid < 112) {
    int j = tid - 109;
    float a = ob2[j];
    for (int k = 0; k < 64; ++k) a += s_h[320 + k] * oW2[k * 3 + j];
    res = tanhf(a) * osc[j]; c = 109 + j;
  }
  if (c >= 0) out[((size_t)b * T_ + t) * 112 + c] = res;
}

// ---------------------------------------------------------------------------
extern "C" void kernel_launch(void* const* d_in, const int* in_sizes, int n_in,
                              void* d_out, int out_size, void* d_ws, size_t ws_size,
                              hipStream_t stream) {
  (void)in_sizes; (void)n_in; (void)out_size; (void)ws_size;
#define P(i) ((const float*)d_in[6 + (i)])
  const float* nmf   = (const float*)d_in[0];
  const float* emo   = (const float*)d_in[1];
  const int*   tok   = (const int*)d_in[2];
  const float* tmask = (const float*)d_in[3];
  const float* phonf = (const float*)d_in[4];
  const float* tfeat = (const float*)d_in[5];
  float* out = (float*)d_out;

  const size_t MB = 1u << 20;
  char* ws = (char*)d_ws;
  bf16*  pk   = (bf16*)ws;                    // packed bf16 weights (~2.4 MB)
  float* sm   = (float*)(ws + 4 * MB);        // small vectors
  bf16*  x0   = (bf16*)(ws + 8 * MB);         // [R,192] bf16  (24 MiB)
  float* giF  = (float*)(ws + 40 * MB);       // [T,16,576]    (144 MiB)
  float* giB  = (float*)(ws + 184 * MB);      // [T,16,576]    (144 MiB)
  bf16*  x1   = (bf16*)(ws + 328 * MB);       // [T,16,384] bf16 (48 MiB)
  float* x2   = (float*)(ws + 376 * MB);      // [T,16,384] f32  (96 MiB)
  bf16*  xln  = (bf16*)(ws + 184 * MB);       // reuse giB
  float* hh   = (float*)(ws + 40 * MB);       // reuse giF

  float* femS = sm;
  float* femT = sm + 3072;
  float* ftxS = sm + 6144;
  float* ftxT = sm + 9216;
  float* amps = sm + 12288;
  float* hbias = sm + 12640;

  bf16* pWih0f = pk;
  bf16* pWih0b = pk + 110592;
  bf16* pWhh0f = pk + 221184;
  bf16* pWhh0b = pk + 331776;
  bf16* pWih1f = pk + 442368;
  bf16* pWih1b = pk + 663552;
  bf16* pWhh1f = pk + 884736;
  bf16* pWhh1b = pk + 995328;
  bf16* pW1    = pk + 1105920;

  auto packL = [&](const float* W, bf16* dst, int K, int N) {
    size_t total = (size_t)K * N;
    pack_b<<<dim3((unsigned)((total + 255) / 256)), dim3(256), 0, stream>>>(W, dst, K, N);
  };
  packL(P(63), pWih0f, 192, 576);
  packL(P(67), pWih0b, 192, 576);
  packL(P(64), pWhh0f, 192, 576);
  packL(P(68), pWhh0b, 192, 576);
  packL(P(71), pWih1f, 384, 576);
  packL(P(75), pWih1b, 384, 576);
  packL(P(72), pWhh1f, 192, 576);
  packL(P(76), pWhh1b, 192, 576);
  packL(P(40), pW1,           384, 192);   // expr_W1  (n-tiles 0..11)
  packL(P(44), pW1 + 73728,   384, 64);    // jaw_W1   (12..15)
  packL(P(48), pW1 + 98304,   384, 64);    // eye_W1   (16..19)
  packL(P(52), pW1 + 122880,  384, 64);    // ori_W1   (20..23)
  pack_hbias<<<1, 384, 0, stream>>>(P(41), P(45), P(49), P(53), hbias);

  prep_small<<<16, 192, 0, stream>>>(
      emo, tok, tmask, phonf,
      P(0), P(1), P(2), P(3), P(4), P(5), P(6),
      P(11), P(12), P(13), P(14),
      P(15), P(16), P(17), P(18), P(19), P(20),
      P(21), P(22), P(23), P(24), P(25), P(26), P(27), P(28), P(29),
      P(30), P(31), P(32), P(33), P(34), P(35), P(36), P(37),
      femS, femT, ftxS, ftxT, amps);

  nmf_film<<<(unsigned)(B_ * T_), 192, 0, stream>>>(
      nmf, tfeat, P(7), P(8), P(9), P(10), femS, femT, ftxS, ftxT, x0);

  // layer 0 input projections (both directions), then recurrent scan
  wmma_gemm<<<dim3(1024, 9), 128, 0, stream>>>(x0, 192, pWih0f, P(65), giF, 576, 192, 0);
  wmma_gemm<<<dim3(1024, 9), 128, 0, stream>>>(x0, 192, pWih0b, P(69), giB, 576, 192, 0);

  const int SCAN_LDS = 221184 + 12288 + 6144 + 2 * 36864;  // 313344
  (void)hipFuncSetAttribute(reinterpret_cast<const void*>(gru_scan),
                            hipFuncAttributeMaxDynamicSharedMemorySize, SCAN_LDS);
  gru_scan<<<2, 192, SCAN_LDS, stream>>>(pWhh0f, pWhh0b, P(66), P(70), giF, giB,
                                         x1, nullptr, T_);

  // layer 1
  wmma_gemm<<<dim3(1024, 9), 128, 0, stream>>>(x1, 384, pWih1f, P(73), giF, 576, 384, 0);
  wmma_gemm<<<dim3(1024, 9), 128, 0, stream>>>(x1, 384, pWih1b, P(77), giB, 576, 384, 0);
  gru_scan<<<2, 192, SCAN_LDS, stream>>>(pWhh1f, pWhh1b, P(74), P(78), giF, giB,
                                         nullptr, x2, T_);

  // post-LN + fused head hidden GEMM (gelu epilogue) + output heads
  post_ln<<<(unsigned)(B_ * T_), 192, 0, stream>>>(x2, P(38), P(39), xln);
  wmma_gemm<<<dim3(1024, 6), 128, 0, stream>>>(xln, 384, pW1, hbias, hh, 384, 384, 1);
  final_heads<<<(unsigned)(B_ * T_), 128, 0, stream>>>(
      hh, nmf, amps, P(42), P(43), P(46), P(47), P(56),
      P(50), P(51), P(57), P(54), P(55), P(58),
      P(61), P(62), P(59), P(60), out);
#undef P
}
